// Attention_16982300688693
// MI455X (gfx1250) — compile-verified, matrix operand order checked by reading, not run on hardware
//
#include <hip/hip_runtime.h>
#include <hip/hip_bf16.h>

// ---------------------------------------------------------------------------
// Problem constants: B=8, N=1025, DIM=768, H=12, D=64, rope Q=16
// ---------------------------------------------------------------------------
#define BATCH 8
#define SEQ   1025
#define SEQP  1056                 // SEQ padded to multiple of 32 (for V^T)
#define DIMM  768
#define HEADS 12
#define HDIM  64
#define MTOK  (BATCH * SEQ)        // 8200 token rows
#define SCALE_ATTN 0.125f          // 64^-0.5

typedef __attribute__((ext_vector_type(16))) __bf16 v16bf;
typedef __attribute__((ext_vector_type(8)))  __bf16 v8bf;
typedef __attribute__((ext_vector_type(8)))  float  v8f;

static __device__ __forceinline__ v16bf combine16(v8bf lo, v8bf hi) {
    union { v16bf v; v8bf h[2]; } u;
    u.h[0] = lo; u.h[1] = hi;
    return u.v;
}

#define WMMA_BF16(A_, B_, C_) \
    __builtin_amdgcn_wmma_f32_16x16x32_bf16(false, (A_), false, (B_), (short)0, (C_), false, false)

// ---------------------------------------------------------------------------
// LDS addressing + gfx1250 async global->LDS copy (ASYNCcnt-tracked).
// ---------------------------------------------------------------------------
typedef __attribute__((address_space(3))) const void lds_cv_t;
static __device__ __forceinline__ unsigned lds_off_u32(const void* p) {
    return (unsigned)(unsigned long long)(lds_cv_t*)p;
}
static __device__ __forceinline__ void async_ld_b128(unsigned lds_byte_off,
                                                     const void* gptr) {
    asm volatile("global_load_async_to_lds_b128 %0, %1, off"
                 :: "v"(lds_byte_off),
                    "v"((unsigned long long)gptr)
                 : "memory");
}
static __device__ __forceinline__ void wait_async0() {
    asm volatile("s_wait_asynccnt 0" ::: "memory");
}

// ---------------------------------------------------------------------------
// Tensor Data Mover: 2D tile DMA global->LDS with HW LDS padding + OOB zero
// fill. Descriptor bitfields per CDNA5 ISA.
// This toolchain's builtin takes 6 args: (g0 x4, g1 x8, x4, x4, x8, cpol).
// ---------------------------------------------------------------------------
#if defined(__has_builtin)
#  if __has_builtin(__builtin_amdgcn_tensor_load_to_lds)
#    define HAVE_TDM 1
#  endif
#endif
#ifndef HAVE_TDM
#  define HAVE_TDM 0
#endif

#if HAVE_TDM
typedef __attribute__((ext_vector_type(4))) unsigned tdm_g0_t;
typedef __attribute__((ext_vector_type(8))) int      tdm_g1_t;
typedef __attribute__((ext_vector_type(4))) int      tdm_g23_t;

// 2-byte elements. pad_interval/pad_amount are the encoded ISA fields.
static __device__ __forceinline__ void tdm_load_2d_bf16(
    unsigned lds_byte_off, const void* gaddr,
    unsigned tensor_d0, unsigned tensor_d1,   // tensor extent (elems, rows)
    unsigned tile_d0, unsigned tile_d1,       // tile extent   (elems, rows)
    unsigned stride_elems,                    // row stride in elements
    unsigned pad_interval, unsigned pad_amount) {
    const unsigned long long ga = (unsigned long long)gaddr;
    tdm_g0_t g0;
    g0[0] = 1u;                                         // count=1, user D#
    g0[1] = lds_byte_off;                               // lds_addr
    g0[2] = (unsigned)(ga & 0xffffffffu);               // global_addr lo
    g0[3] = (unsigned)((ga >> 32) & 0x1ffffffu)         // global_addr[56:32]
          | (2u << 30);                                 // type = 2 ("image")
    tdm_g1_t g1;
    g1[0] = (int)((1u << 16)                            // data_size = 2B
                | (1u << 20)                            // pad_enable
                | (pad_interval << 22)                  // dwords before pad
                | (pad_amount << 25));                  // pad dwords - 1
    g1[1] = (int)((tensor_d0 & 0xffffu) << 16);         // dim0[15:0] @ b63:48
    g1[2] = (int)(((tensor_d0 >> 16) & 0xffffu)         // dim0[31:16]
                | ((tensor_d1 & 0xffffu) << 16));       // dim1[15:0]
    g1[3] = (int)(((tensor_d1 >> 16) & 0xffffu)         // dim1[31:16]
                | (tile_d0 << 16));                     // tile_dim0
    g1[4] = (int)(tile_d1 & 0xffffu);                   // tile_dim1, dim2=0
    g1[5] = (int)stride_elems;                          // dim0_stride[31:0]
    g1[6] = 0;                                          // stride hi, ts1 lo
    g1[7] = 0;
    const tdm_g23_t gz4 = {0, 0, 0, 0};                 // 2D: groups 2/3 idle
    const tdm_g1_t  gz8 = {0, 0, 0, 0, 0, 0, 0, 0};     // unused extra group
    __builtin_amdgcn_tensor_load_to_lds(g0, g1, gz4, gz4, gz8, 0);
}
#endif

// ---------------------------------------------------------------------------
// fp32 -> bf16 elementwise convert
// ---------------------------------------------------------------------------
__global__ void cvt_f32_to_bf16(const float* __restrict__ in,
                                __bf16* __restrict__ out, int n) {
    int i = blockIdx.x * blockDim.x + threadIdx.x;
    if (i < n) out[i] = (__bf16)in[i];
}

// ---------------------------------------------------------------------------
// GEMM: C[M,N] = A[M,K](bf16) * B[N,K](bf16)^T + bias[N], C fp32.
// 128x128x64 tiles, 8 waves (2x4). Staging: TDM double-buffered DMA
// (wave0 -> A tile, wave1 -> B tile; HW pads LDS rows to 72 halves and
// zero-fills OOB tail rows). 16 v_wmma per wave per K-slab.
// ---------------------------------------------------------------------------
__global__ __launch_bounds__(256)
void gemm_bf16_wmma(const __bf16* __restrict__ A, const __bf16* __restrict__ B,
                    const float* __restrict__ bias, float* __restrict__ C,
                    int M, int N, int K) {
    constexpr int BM = 128, BK = 64;
    constexpr int LDT = BK + 8;                  // 72 halves = 144B (16B mult)
    __shared__ __align__(16) __bf16 As[2][BM * LDT];
    __shared__ __align__(16) __bf16 Bs[2][BM * LDT];

    const int tid  = threadIdx.x;
    const int wave = tid >> 5;
    const int lane = tid & 31;
    const int r    = lane & 15;
    const int hh   = lane >> 4;
    const int wm   = wave >> 2;                  // 0..1  -> 64-row patch
    const int wn   = wave & 3;                   // 0..3  -> 32-col patch
    const int rowBase = blockIdx.x * BM;
    const int colBase = blockIdx.y * BM;

    v8f acc[4][2];
    for (int i = 0; i < 4; ++i)
        for (int j = 0; j < 2; ++j) acc[i][j] = {};

    auto stage = [&](int buf, int k0) {
#if HAVE_TDM
        // tile rows = 64 halves = 32 dwords -> pad_interval=4 (every 32
        // dwords), pad_amount=3 (+4 dwords) => LDS row stride = 72 halves.
        if (wave == 0)
            tdm_load_2d_bf16(lds_off_u32(&As[buf][0]),
                             A + (size_t)rowBase * K + k0,
                             BK, (unsigned)(M - rowBase),  // OOB rows -> 0
                             BK, BM, (unsigned)K, 4u, 3u);
        else if (wave == 1)
            tdm_load_2d_bf16(lds_off_u32(&Bs[buf][0]),
                             B + (size_t)colBase * K + k0,
                             BK, (unsigned)(N - colBase),
                             BK, BM, (unsigned)K, 4u, 3u);
#else
        for (int p = 0; p < 4; ++p) {
            const int slot = p * 256 + tid;
            const int rr = slot >> 3;
            const int cc = (slot & 7) * 8;
            async_ld_b128(lds_off_u32(&As[buf][rr * LDT + cc]),
                          A + (size_t)(rowBase + rr) * K + k0 + cc);
            async_ld_b128(lds_off_u32(&Bs[buf][rr * LDT + cc]),
                          B + (size_t)(colBase + rr) * K + k0 + cc);
        }
#endif
    };
    auto wait_stage = [&]() {
#if HAVE_TDM
        __builtin_amdgcn_s_wait_tensorcnt(0);
#else
        wait_async0();
#endif
    };

    const int NK = K / BK;                       // 768/64 = 12
    stage(0, 0);
    wait_stage();
    __syncthreads();

    for (int s = 0; s < NK; ++s) {
        const int cur = s & 1;
        if (s + 1 < NK) stage(cur ^ 1, (s + 1) * BK);   // DMA next slab

        for (int kk = 0; kk < BK; kk += 32) {
            v16bf af[4], bfv[2];
            for (int i = 0; i < 4; ++i) {
                const __bf16* base = &As[cur][(wm * 64 + i * 16 + r) * LDT + kk];
                af[i] = combine16(*(const v8bf*)(base + hh * 8),
                                  *(const v8bf*)(base + 16 + hh * 8));
            }
            for (int j = 0; j < 2; ++j) {
                const __bf16* base = &Bs[cur][(wn * 32 + j * 16 + r) * LDT + kk];
                bfv[j] = combine16(*(const v8bf*)(base + hh * 8),
                                   *(const v8bf*)(base + 16 + hh * 8));
            }
            for (int i = 0; i < 4; ++i)
                for (int j = 0; j < 2; ++j)
                    acc[i][j] = WMMA_BF16(af[i], bfv[j], acc[i][j]);
        }
        wait_stage();                            // next slab resident in LDS
        __syncthreads();                         // everyone done reading cur
    }

    for (int i = 0; i < 4; ++i) {
        const int row0 = rowBase + wm * 64 + i * 16;
        for (int j = 0; j < 2; ++j) {
            const int col = colBase + wn * 32 + j * 16 + r;
            const float bv = bias ? bias[col] : 0.0f;
            for (int e = 0; e < 8; ++e) {
                const int row = row0 + e + 8 * hh;
                if (row < M) C[(size_t)row * N + col] = acc[i][j][e] + bv;
            }
        }
    }
}

// ---------------------------------------------------------------------------
// Per-head LayerNorm + RoPE + layout transform. One wave per (b,h,n),
// n in [0,SEQP): n >= SEQ only zero-fills the V^T pad (NaN-safe attention).
// q/k: bf16 [b,h,n,d].  v: bf16 transposed [b,h,d,n] with n padded to SEQP.
// ---------------------------------------------------------------------------
__global__ __launch_bounds__(256)
void ln_rope_kernel(const float* __restrict__ qkv,
                    const float* __restrict__ cy, const float* __restrict__ sy,
                    const float* __restrict__ cx, const float* __restrict__ sx,
                    const float* __restrict__ qg, const float* __restrict__ qb,
                    const float* __restrict__ kg, const float* __restrict__ kb,
                    __bf16* __restrict__ qo, __bf16* __restrict__ ko,
                    __bf16* __restrict__ vt) {
    const int wid  = blockIdx.x * (blockDim.x >> 5) + (threadIdx.x >> 5);
    const int lane = threadIdx.x & 31;
    const int total = BATCH * HEADS * SEQP;
    if (wid >= total) return;                     // wave-uniform exit

    const int b   = wid / (HEADS * SEQP);
    const int rem = wid - b * (HEADS * SEQP);
    const int h   = rem / SEQP;
    const int n   = rem - h * SEQP;
    const int bh  = b * HEADS + h;

    const size_t vt0 = ((size_t)bh * HDIM + lane) * SEQP + n;
    const size_t vt1 = ((size_t)bh * HDIM + lane + 32) * SEQP + n;
    if (n >= SEQ) {                               // zero the V^T pad columns
        vt[vt0] = (__bf16)0.0f;
        vt[vt1] = (__bf16)0.0f;
        return;
    }

    const size_t tok = (size_t)b * SEQ + n;
    const float* base = qkv + tok * (3 * DIMM) + h * HDIM;
    const size_t obase = ((size_t)bh * SEQ + n) * HDIM;

    auto lnrope = [&](const float* p, const float* g, const float* be,
                      __bf16* out) {
        const float a0 = p[lane], a1 = p[lane + 32];
        float s = a0 + a1;
        for (int msk = 1; msk < 32; msk <<= 1) s += __shfl_xor(s, msk, 32);
        const float mean = s * (1.0f / 64.0f);
        const float e0 = a0 - mean, e1 = a1 - mean;
        float vv = e0 * e0 + e1 * e1;
        for (int msk = 1; msk < 32; msk <<= 1) vv += __shfl_xor(vv, msk, 32);
        const float rstd = rsqrtf(vv * (1.0f / 64.0f) + 1e-5f);
        float y0 = e0 * rstd * g[lane]      + be[lane];
        float y1 = e1 * rstd * g[lane + 32] + be[lane + 32];
        if (n > 0) {                              // RoPE on tokens 1..N-1
            const int i = lane & 15;
            const size_t ro = (size_t)(n - 1) * 16 + i;
            const float p0 = __shfl_xor(y0, 16, 32);
            const float p1 = __shfl_xor(y1, 16, 32);
            const float sgn = (lane < 16) ? -1.0f : 1.0f;
            y0 = y0 * cy[ro] + sgn * sy[ro] * p0; // y1/y2 halves
            y1 = y1 * cx[ro] + sgn * sx[ro] * p1; // x1/x2 halves
        }
        out[obase + lane]      = (__bf16)y0;
        out[obase + lane + 32] = (__bf16)y1;
    };
    lnrope(base,        qg, qb, qo);
    lnrope(base + DIMM, kg, kb, ko);
    vt[vt0] = (__bf16)base[2 * DIMM + lane];
    vt[vt1] = (__bf16)base[2 * DIMM + lane + 32];
}

// ---------------------------------------------------------------------------
// Flash attention. Grid: (B*H, ceil(N/128)), 8 waves x 16 query rows.
// K and V^T staged by async double-buffered LDS copies; S via 4 WMMAs,
// online softmax (16-lane xor reductions), O += P*V via 4 WMMAs.
// ---------------------------------------------------------------------------
__global__ __launch_bounds__(256)
void attn_kernel(const __bf16* __restrict__ q, const __bf16* __restrict__ k,
                 const __bf16* __restrict__ vt, __bf16* __restrict__ out) {
    constexpr int LDK = HDIM + 8;                 // 72 halves = 144B
    constexpr int LDV = 48;                       // 96B stride
    __shared__ __align__(16) __bf16 Ks[2][32 * LDK];
    __shared__ __align__(16) __bf16 Vts[2][HDIM * LDV];
    __shared__ __align__(16) __bf16 Ps[8][16 * LDV];

    const int bh = blockIdx.x;
    const int b = bh / HEADS, h = bh % HEADS;
    const int tid = threadIdx.x, wave = tid >> 5, lane = tid & 31;
    const int r = lane & 15, hh = lane >> 4;
    const int qrow0 = blockIdx.y * 128 + wave * 16;

    const __bf16* qbase  = q  + (size_t)bh * SEQ * HDIM;
    const __bf16* kbase  = k  + (size_t)bh * SEQ * HDIM;
    const __bf16* vtbase = vt + (size_t)bh * HDIM * SEQP;

    int qr = qrow0 + r; if (qr >= SEQ) qr = SEQ - 1;   // clamp; never stored
    const __bf16* qp = qbase + (size_t)qr * HDIM;
    const v16bf qf0 = combine16(*(const v8bf*)(qp + hh * 8),
                                *(const v8bf*)(qp + 16 + hh * 8));
    const v16bf qf1 = combine16(*(const v8bf*)(qp + 32 + hh * 8),
                                *(const v8bf*)(qp + 48 + hh * 8));

    v8f o[4];
    for (int t = 0; t < 4; ++t) o[t] = {};
    float mrun[8], lrun[8];
    for (int e = 0; e < 8; ++e) { mrun[e] = -1e30f; lrun[e] = 0.0f; }

    auto stageKV = [&](int buf, int key0) {
        {   // K[32 x 64] row-major: 256 x 16B transfers (tail reads padded)
            const int row = tid >> 3, cc = (tid & 7) * 8;
            async_ld_b128(lds_off_u32(&Ks[buf][row * LDK + cc]),
                          kbase + (size_t)(key0 + row) * HDIM + cc);
        }
        {   // V^T[64 x 32]: rows d, key columns; pad columns are zeroed
            const int d = tid >> 2, cc = (tid & 3) * 8;
            async_ld_b128(lds_off_u32(&Vts[buf][d * LDV + cc]),
                          vtbase + (size_t)d * SEQP + key0 + cc);
        }
    };

    const int KB = (SEQ + 31) / 32;               // 33 key blocks
    stageKV(0, 0);
    wait_async0();
    __syncthreads();

    for (int kb = 0; kb < KB; ++kb) {
        const int cur = kb & 1;
        const int key0 = kb * 32;
        if (kb + 1 < KB) stageKV(cur ^ 1, (kb + 1) * 32);

        // S = (Q * K^T) * scale  (two 16x16 key tiles)
        v8f s[2];
        for (int j = 0; j < 2; ++j) {
            const __bf16* kp = &Ks[cur][(j * 16 + r) * LDK];
            const v16bf b0 = combine16(*(const v8bf*)(kp + hh * 8),
                                       *(const v8bf*)(kp + 16 + hh * 8));
            const v16bf b1 = combine16(*(const v8bf*)(kp + 32 + hh * 8),
                                       *(const v8bf*)(kp + 48 + hh * 8));
            v8f t = {};
            t = WMMA_BF16(qf0, b0, t);
            t = WMMA_BF16(qf1, b1, t);
            const int key = key0 + j * 16 + r;
            const bool bad = key >= SEQ;
            for (int e = 0; e < 8; ++e)
                t[e] = bad ? -1e30f : t[e] * SCALE_ATTN;
            s[j] = t;
        }

        // online softmax: row = e + 8*hh matches accumulator layout
        float alpha[8];
        for (int e = 0; e < 8; ++e) {
            float t = fmaxf(s[0][e], s[1][e]);
            for (int msk = 1; msk < 16; msk <<= 1)
                t = fmaxf(t, __shfl_xor(t, msk, 32));
            const float mn = fmaxf(mrun[e], t);
            alpha[e] = __expf(mrun[e] - mn);
            mrun[e] = mn;
        }
        for (int j = 0; j < 2; ++j)
            for (int e = 0; e < 8; ++e)
                s[j][e] = __expf(s[j][e] - mrun[e]);
        for (int e = 0; e < 8; ++e) {
            float t = s[0][e] + s[1][e];
            for (int msk = 1; msk < 16; msk <<= 1) t += __shfl_xor(t, msk, 32);
            lrun[e] = lrun[e] * alpha[e] + t;
        }

        // P through wave-private LDS scratch: C-frag -> A-frag relayout
        __bf16* pw = &Ps[wave][0];
        for (int j = 0; j < 2; ++j)
            for (int e = 0; e < 8; ++e)
                pw[(e + 8 * hh) * LDV + j * 16 + r] = (__bf16)s[j][e];
        const __bf16* pr = &Ps[wave][r * LDV];
        const v16bf pf = combine16(*(const v8bf*)(pr + hh * 8),
                                   *(const v8bf*)(pr + 16 + hh * 8));

        for (int t = 0; t < 4; ++t) {
            for (int e = 0; e < 8; ++e) o[t][e] *= alpha[e];
            const __bf16* vp = &Vts[cur][(t * 16 + r) * LDV];
            const v16bf vf = combine16(*(const v8bf*)(vp + hh * 8),
                                       *(const v8bf*)(vp + 16 + hh * 8));
            o[t] = WMMA_BF16(pf, vf, o[t]);
        }

        wait_async0();                            // next K/V block resident
        __syncthreads();
    }

    for (int t = 0; t < 4; ++t)
        for (int e = 0; e < 8; ++e) {
            const int row = qrow0 + e + 8 * hh;
            if (row < SEQ) {
                const float val = o[t][e] / lrun[e];
                out[((size_t)b * SEQ + row) * DIMM + h * HDIM + t * 16 + r] =
                    (__bf16)val;
            }
        }
}

// ---------------------------------------------------------------------------
// Launcher
// ---------------------------------------------------------------------------
extern "C" void kernel_launch(void* const* d_in, const int* in_sizes, int n_in,
                              void* d_out, int out_size, void* d_ws, size_t ws_size,
                              hipStream_t stream) {
    const float* x      = (const float*)d_in[0];
    const float* cy     = (const float*)d_in[1];
    const float* sy     = (const float*)d_in[2];
    const float* cx     = (const float*)d_in[3];
    const float* sx     = (const float*)d_in[4];
    const float* qkv_w  = (const float*)d_in[5];
    const float* qkv_b  = (const float*)d_in[6];
    const float* proj_w = (const float*)d_in[7];
    const float* proj_b = (const float*)d_in[8];
    const float* q_g    = (const float*)d_in[9];
    const float* q_b    = (const float*)d_in[10];
    const float* k_g    = (const float*)d_in[11];
    const float* k_b    = (const float*)d_in[12];
    float* out = (float*)d_out;

    char* ws = (char*)d_ws;
    size_t off = 0;
    auto bump = [&](size_t bytes) -> void* {
        void* p = ws + off;
        off += (bytes + 255) & ~(size_t)255;
        return p;
    };
    const size_t PAD = 256 * 1024;   // OOB slack for unguarded async tail reads
    __bf16* x_bf  = (__bf16*)bump((size_t)MTOK * DIMM * 2 + PAD);
    __bf16* wq_bf = (__bf16*)bump((size_t)3 * DIMM * DIMM * 2);
    __bf16* wp_bf = (__bf16*)bump((size_t)DIMM * DIMM * 2);
    float*  qkv_f = (float*) bump((size_t)MTOK * 3 * DIMM * 4);
    __bf16* q_bf  = (__bf16*)bump((size_t)BATCH * HEADS * SEQ * HDIM * 2);
    __bf16* k_bf  = (__bf16*)bump((size_t)BATCH * HEADS * SEQ * HDIM * 2 + PAD);
    __bf16* vt_bf = (__bf16*)bump((size_t)BATCH * HEADS * HDIM * SEQP * 2);
    __bf16* ao_bf = (__bf16*)bump((size_t)MTOK * DIMM * 2 + PAD);
    (void)ws_size; (void)in_sizes; (void)n_in; (void)out_size;

    // 1. converts
    {
        const int nx = MTOK * DIMM, nq = 3 * DIMM * DIMM, np = DIMM * DIMM;
        cvt_f32_to_bf16<<<(nx + 255) / 256, 256, 0, stream>>>(x, x_bf, nx);
        cvt_f32_to_bf16<<<(nq + 255) / 256, 256, 0, stream>>>(qkv_w, wq_bf, nq);
        cvt_f32_to_bf16<<<(np + 255) / 256, 256, 0, stream>>>(proj_w, wp_bf, np);
    }
    // 2. QKV GEMM: [8200,768] x [2304,768]^T + b -> fp32
    {
        dim3 grid((MTOK + 127) / 128, (3 * DIMM) / 128);
        gemm_bf16_wmma<<<grid, 256, 0, stream>>>(x_bf, wq_bf, qkv_b, qkv_f,
                                                 MTOK, 3 * DIMM, DIMM);
    }
    // 3. LayerNorm + RoPE, q/k -> [b,h,n,d], v -> transposed padded [b,h,d,n]
    {
        const int waves = BATCH * HEADS * SEQP;         // 101376
        ln_rope_kernel<<<(waves + 7) / 8, 256, 0, stream>>>(
            qkv_f, cy, sy, cx, sx, q_g, q_b, k_g, k_b, q_bf, k_bf, vt_bf);
    }
    // 4. Flash attention
    {
        dim3 grid(BATCH * HEADS, (SEQ + 127) / 128);
        attn_kernel<<<grid, 256, 0, stream>>>(q_bf, k_bf, vt_bf, ao_bf);
    }
    // 5. Out projection: [8200,768] x [768,768]^T + b -> d_out fp32
    {
        dim3 grid((MTOK + 127) / 128, DIMM / 128);
        gemm_bf16_wmma<<<grid, 256, 0, stream>>>(ao_bf, wp_bf, proj_b, out,
                                                 MTOK, DIMM, DIMM);
    }
}
